// AdvancedRQVAE_89850715833214
// MI455X (gfx1250) — compile-verified, matrix operand order checked by reading, not run on hardware
//
#include <hip/hip_runtime.h>
#include <math.h>
#include <stdint.h>

#define BATCH 32768

typedef __attribute__((ext_vector_type(16))) __bf16 v16bf;
typedef __attribute__((ext_vector_type(8)))  float  v8f;

__device__ __forceinline__ float gelu_exact(float x) {
  return 0.5f * x * (1.0f + erff(x * 0.70710678118654752440f));
}

// async copy 16B/lane from global -> LDS, tracked by ASYNCcnt (gfx1250)
__device__ __forceinline__ void async_b128_to_lds(const float* g, void* lds) {
  unsigned l = (unsigned)(uintptr_t)lds;  // low 32 bits of flat shared ptr = LDS byte offset
  asm volatile("global_load_async_to_lds_b128 %0, %1, off" ::"v"(l), "v"(g) : "memory");
}
__device__ __forceinline__ void wait_async0() {
  asm volatile("s_wait_asynccnt 0" ::: "memory");
}

// ------------------------------------------------------------------
// weight packing: w [K, N] fp32 row-major  ->  Wt [N, Kpad] bf16 (zero pad)
// ------------------------------------------------------------------
__global__ void pack_wt_kernel(const float* __restrict__ w, __bf16* __restrict__ wt,
                               int N, int K, int Kpad) {
  int i = blockIdx.x * blockDim.x + threadIdx.x;
  if (i >= N * Kpad) return;
  int n = i / Kpad, k = i - n * Kpad;
  float v = (k < K) ? w[(size_t)k * N + n] : 0.0f;
  wt[i] = (__bf16)v;
}

__global__ void pack_cb_kernel(const float* __restrict__ w, __bf16* __restrict__ wt, int total) {
  int i = blockIdx.x * blockDim.x + threadIdx.x;
  if (i < total) wt[i] = (__bf16)w[i];
}

__global__ void cb_norm_kernel(const float* __restrict__ w, float* __restrict__ wn, int Kl) {
  int n = blockIdx.x * blockDim.x + threadIdx.x;
  if (n >= Kl) return;
  const float* row = w + (size_t)n * 256;
  float s = 0.f;
  for (int d = 0; d < 256; ++d) { float v = row[d]; s += v * v; }
  wn[n] = s;
}

__global__ void fill_fused_kernel(const float* __restrict__ dm, float* __restrict__ fused) {
  int i = blockIdx.x * blockDim.x + threadIdx.x;
  int b = i >> 5, c = i & 31;
  fused[(size_t)b * 544 + 512 + c] = (c == 0) ? dm[b] : 0.0f;
}

__global__ void write_mask_kernel(float* __restrict__ m, const int* __restrict__ act) {
  int i = threadIdx.x;
  if (i < 4) m[i] = (i < *act) ? 1.0f : 0.0f;
}

// ------------------------------------------------------------------
// Fused GEMM: C[m, coloff+n] = epilogue( A[m,:] @ Wt[n,:] + bias[n] (+resid) )
//   block: 32 rows (2 M-tiles) x N cols (N = TPW*128), 8 waves.
//   A tile double-buffered in LDS via global_load_async_to_lds_b128.
//   EPI: 0=none 1=gelu 2=ln 3=ln->gelu 4=gelu->ln 5=l2normalize
// ------------------------------------------------------------------
template <int TPW, int EPI>
__global__ __launch_bounds__(256) void gemm_epi(
    const float* __restrict__ A, int lda,
    const __bf16* __restrict__ Wt,
    const float* __restrict__ bias,
    const float* __restrict__ lng, const float* __restrict__ lnb,
    const float* __restrict__ resid, int ldr,
    float* __restrict__ C, long ldc, long coloff, int K) {
  constexpr int N = TPW * 128;
  __shared__ __align__(16) float Afs[2][32][36];  // fp32 A slab, dbl-buffered, +pad
  __shared__ float Cs[16][N + 4];

  const int tid  = threadIdx.x;
  const int wave = tid >> 5;
  const int lane = tid & 31;
  const int m    = lane & 15;
  const int hk   = (lane >> 4) * 8;   // half-of-wave K base
  const int mb   = (lane >> 4) * 8;   // C-fragment row base
  const size_t row0 = (size_t)blockIdx.x * 32;
  const int sr = tid >> 3;            // stage row 0..31
  const int sc = (tid & 7) * 4;       // stage col (floats)

  v8f acc[2][TPW];
#pragma unroll
  for (int mt = 0; mt < 2; ++mt)
#pragma unroll
    for (int j = 0; j < TPW; ++j)
#pragma unroll
      for (int r = 0; r < 8; ++r) acc[mt][j][r] = 0.0f;

  const int nsteps = K >> 5;
  async_b128_to_lds(A + (row0 + sr) * lda + sc, &Afs[0][sr][sc]);
  wait_async0();
  __syncthreads();

  for (int s = 0; s < nsteps; ++s) {
    const int k0 = s << 5;
    const int cur = s & 1;
    if (s + 1 < nsteps)
      async_b128_to_lds(A + (row0 + sr) * lda + k0 + 32 + sc, &Afs[cur ^ 1][sr][sc]);

    v16bf a[2];
#pragma unroll
    for (int mt = 0; mt < 2; ++mt) {
      const float* p0 = &Afs[cur][mt * 16 + m][hk];
      const float* p1 = &Afs[cur][mt * 16 + m][16 + hk];
#pragma unroll
      for (int i = 0; i < 8; ++i) { a[mt][i] = (__bf16)p0[i]; a[mt][i + 8] = (__bf16)p1[i]; }
    }
#pragma unroll
    for (int j = 0; j < TPW; ++j) {
      int t = wave + 8 * j;
      const __bf16* bp = Wt + (size_t)(t * 16 + m) * K + k0 + hk;
      v16bf b;
#pragma unroll
      for (int i = 0; i < 8; ++i) { b[i] = bp[i]; b[i + 8] = bp[16 + i]; }
      acc[0][j] = __builtin_amdgcn_wmma_f32_16x16x32_bf16(
          false, a[0], false, b, (short)0, acc[0][j], false, false);
      acc[1][j] = __builtin_amdgcn_wmma_f32_16x16x32_bf16(
          false, a[1], false, b, (short)0, acc[1][j], false, false);
    }
    if (s + 1 < nsteps) wait_async0();
    __syncthreads();
  }

  // two-pass epilogue over the two 16-row M-tiles through one LDS C tile
  const int er = tid >> 4, ec = tid & 15;
  constexpr int CPT = N / 16;
#pragma unroll
  for (int mt = 0; mt < 2; ++mt) {
#pragma unroll
    for (int j = 0; j < TPW; ++j) {
      int ncol = (wave + 8 * j) * 16 + m;
      float bv = bias ? bias[ncol] : 0.0f;
#pragma unroll
      for (int r = 0; r < 8; ++r) {
        float v = acc[mt][j][r] + bv;
        if (resid) v += resid[(row0 + mt * 16 + mb + r) * (size_t)ldr + ncol];
        Cs[mb + r][ncol] = v;
      }
    }
    __syncthreads();
    float* crow = Cs[er];
    float* outp = C + (row0 + mt * 16 + er) * ldc + coloff;

    if constexpr (EPI == 0 || EPI == 1) {
#pragma unroll
      for (int i = 0; i < CPT; ++i) {
        int c = ec + 16 * i;
        float v = crow[c];
        if constexpr (EPI == 1) v = gelu_exact(v);
        outp[c] = v;
      }
    } else if constexpr (EPI == 5) {
      float ss = 0.f;
#pragma unroll
      for (int i = 0; i < CPT; ++i) { float v = crow[ec + 16 * i]; ss += v * v; }
      for (int msk = 8; msk; msk >>= 1) ss += __shfl_xor(ss, msk, 16);
      float scale = 1.0f / fmaxf(sqrtf(ss), 1e-12f);
#pragma unroll
      for (int i = 0; i < CPT; ++i) { int c = ec + 16 * i; outp[c] = crow[c] * scale; }
    } else {  // 2, 3, 4 : layernorm variants
      float s = 0.f, ss = 0.f;
#pragma unroll
      for (int i = 0; i < CPT; ++i) {
        int c = ec + 16 * i;
        float v = crow[c];
        if constexpr (EPI == 4) { v = gelu_exact(v); crow[c] = v; }
        s += v; ss += v * v;
      }
      for (int msk = 8; msk; msk >>= 1) {
        s  += __shfl_xor(s,  msk, 16);
        ss += __shfl_xor(ss, msk, 16);
      }
      float mean = s / N;
      float var  = ss / N - mean * mean;
      float inv  = rsqrtf(var + 1e-5f);
#pragma unroll
      for (int i = 0; i < CPT; ++i) {
        int c = ec + 16 * i;
        float v = (crow[c] - mean) * inv * lng[c] + lnb[c];
        if constexpr (EPI == 3) v = gelu_exact(v);
        outp[c] = v;
      }
    }
    __syncthreads();
  }
}

// ------------------------------------------------------------------
// Residual-quantization level: 32 rows/block; residual fragments live in
// VGPRs; each B fragment feeds two independent WMMA chains; branchless
// running argmin (v_cmp + v_cndmask, no exec-mask branches).
// ------------------------------------------------------------------
__global__ __launch_bounds__(256) void rq_level(
    const float* __restrict__ rin, long ld_in, int do_copy,
    const __bf16* __restrict__ CBt, const float* __restrict__ cbf,
    const float* __restrict__ wn, int Kl,
    float* __restrict__ rcopy,            // stride 1024 (residuals[:,l,:])
    float* __restrict__ qout,             // stride 1024 (quantized[:,l,:])
    float* __restrict__ rnext, long ld_rn,
    float* __restrict__ zq, float* __restrict__ zqst,
    float* __restrict__ sids,
    const int* __restrict__ act, int level) {
  __shared__ __align__(16) float  Rs[32][256];
  __shared__ __align__(16) __bf16 Ab[32][256];
  __shared__ float candv[32][8];
  __shared__ int   candi[32][8];
  __shared__ int   rowidx[32];

  const int tid  = threadIdx.x;
  const int wave = tid >> 5;
  const int lane = tid & 31;
  const int m    = lane & 15;
  const int hb   = (lane >> 4) * 8;
  const size_t row0 = (size_t)blockIdx.x * 32;

  for (int i = tid; i < 32 * 256; i += 256) {
    int r = i >> 8, c = i & 255;
    float v = rin[(row0 + r) * ld_in + c];
    Rs[r][c] = v;
    Ab[r][c] = (__bf16)v;
    if (do_copy) rcopy[(row0 + r) * 1024 + c] = v;
  }
  __syncthreads();

  v16bf af[2][8];  // two 16x256 residual fragments (8 k-steps each)
#pragma unroll
  for (int mt = 0; mt < 2; ++mt)
#pragma unroll
    for (int ks = 0; ks < 8; ++ks) {
      const __bf16* p0 = &Ab[mt * 16 + m][ks * 32 + hb];
      const __bf16* p1 = &Ab[mt * 16 + m][ks * 32 + 16 + hb];
#pragma unroll
      for (int i = 0; i < 8; ++i) { af[mt][ks][i] = p0[i]; af[mt][ks][i + 8] = p1[i]; }
    }

  float minv[2][8]; int mini[2][8];
#pragma unroll
  for (int mt = 0; mt < 2; ++mt)
#pragma unroll
    for (int r = 0; r < 8; ++r) { minv[mt][r] = 3.4e38f; mini[mt][r] = 0x7fffffff; }

  const int ntile = Kl >> 4;
  for (int t = wave; t < ntile; t += 8) {
    v8f acc0, acc1;
#pragma unroll
    for (int r = 0; r < 8; ++r) { acc0[r] = 0.0f; acc1[r] = 0.0f; }
    const __bf16* base = CBt + (size_t)(t * 16 + m) * 256;
#pragma unroll
    for (int ks = 0; ks < 8; ++ks) {
      const __bf16* bp = base + ks * 32 + hb;
      v16bf b;
#pragma unroll
      for (int i = 0; i < 8; ++i) { b[i] = bp[i]; b[i + 8] = bp[16 + i]; }
      acc0 = __builtin_amdgcn_wmma_f32_16x16x32_bf16(
          false, af[0][ks], false, b, (short)0, acc0, false, false);
      acc1 = __builtin_amdgcn_wmma_f32_16x16x32_bf16(
          false, af[1][ks], false, b, (short)0, acc1, false, false);
    }
    const int n = t * 16 + m;
    const float w2 = wn[n];
#pragma unroll
    for (int r = 0; r < 8; ++r) {
      float d0 = fmaf(-2.0f, acc0[r], w2);
      float d1 = fmaf(-2.0f, acc1[r], w2);
      bool b0 = (d0 < minv[0][r]) | ((d0 == minv[0][r]) & (n < mini[0][r]));
      bool b1 = (d1 < minv[1][r]) | ((d1 == minv[1][r]) & (n < mini[1][r]));
      minv[0][r] = b0 ? d0 : minv[0][r];
      mini[0][r] = b0 ? n : mini[0][r];
      minv[1][r] = b1 ? d1 : minv[1][r];
      mini[1][r] = b1 ? n : mini[1][r];
    }
  }
  // reduce 16 candidate columns per row inside each half-wave (branchless)
#pragma unroll
  for (int mt = 0; mt < 2; ++mt)
#pragma unroll
    for (int r = 0; r < 8; ++r) {
      for (int msk = 8; msk; msk >>= 1) {
        float ov = __shfl_xor(minv[mt][r], msk, 16);
        int   oi = __shfl_xor(mini[mt][r], msk, 16);
        bool bt = (ov < minv[mt][r]) | ((ov == minv[mt][r]) & (oi < mini[mt][r]));
        minv[mt][r] = bt ? ov : minv[mt][r];
        mini[mt][r] = bt ? oi : mini[mt][r];
      }
    }
  if (m == 0) {
    int rb = (lane >> 4) * 8;
#pragma unroll
    for (int mt = 0; mt < 2; ++mt)
#pragma unroll
      for (int r = 0; r < 8; ++r) {
        candv[mt * 16 + rb + r][wave] = minv[mt][r];
        candi[mt * 16 + rb + r][wave] = mini[mt][r];
      }
  }
  __syncthreads();
  if (tid < 32) {  // reduce across 8 waves (branchless)
    float bv = candv[tid][0]; int bi = candi[tid][0];
#pragma unroll
    for (int w = 1; w < 8; ++w) {
      float v = candv[tid][w]; int ix = candi[tid][w];
      bool bt = (v < bv) | ((v == bv) & (ix < bi));
      bv = bt ? v : bv;
      bi = bt ? ix : bi;
    }
    rowidx[tid] = bi;
    sids[(row0 + tid) * 4 + level] = (float)bi;
  }
  __syncthreads();

  const int am = (level < *act) ? 1 : 0;
  const int er = tid >> 3, ec = tid & 7;  // 32 rows x 8 threads
  const int idx = rowidx[er];
  const float* qrow = cbf + (size_t)idx * 256;
  const size_t gr = row0 + er;
#pragma unroll
  for (int i = 0; i < 32; ++i) {
    int c = ec + 8 * i;
    float q = qrow[c];
    float r = Rs[er][c];
    qout[gr * 1024 + c] = q;
    rnext[gr * ld_rn + c] = r - q;
    float contrib = am ? q : 0.0f;
    if (level == 0) { zq[gr * 256 + c] = contrib;  zqst[gr * 256 + c] = contrib; }
    else            { zq[gr * 256 + c] += contrib; zqst[gr * 256 + c] += contrib; }
  }
}

// ------------------------------------------------------------------
extern "C" void kernel_launch(void* const* d_in, const int* in_sizes, int n_in,
                              void* d_out, int out_size, void* d_ws, size_t ws_size,
                              hipStream_t stream) {
  (void)in_sizes; (void)n_in; (void)out_size; (void)ws_size;
  auto F = [&](int i) { return (const float*)d_in[i]; };
  const float* text  = F(0);
  const float* image = F(1);
  const float* dmask = F(2);
  const int*   act   = (const int*)d_in[73];

  // ---- workspace carve-up ----
  char* ws = (char*)d_ws;
  size_t off = 0;
  auto alloc_bf = [&](size_t n) { __bf16* p = (__bf16*)(ws + off); off += ((n * 2) + 255) & ~(size_t)255; return p; };
  auto alloc_f  = [&](size_t n) { float*  p = (float*)(ws + off);  off += ((n * 4) + 255) & ~(size_t)255; return p; };

  __bf16* wt_bt_p0 = alloc_bf((size_t)512 * 768);
  __bf16* wt_bt_po = alloc_bf((size_t)256 * 512);
  __bf16* wt_bt_b1 = alloc_bf((size_t)512 * 256);
  __bf16* wt_bt_b2 = alloc_bf((size_t)256 * 512);
  __bf16* wt_bi_p0 = alloc_bf((size_t)512 * 512);
  __bf16* wt_bi_po = alloc_bf((size_t)256 * 512);
  __bf16* wt_bi_b1 = alloc_bf((size_t)512 * 256);
  __bf16* wt_bi_b2 = alloc_bf((size_t)256 * 512);
  __bf16* wt_f_h0  = alloc_bf((size_t)512 * 544);
  __bf16* wt_f_h1  = alloc_bf((size_t)512 * 512);
  __bf16* wt_f_o   = alloc_bf((size_t)256 * 512);
  __bf16* wt_dt_h0 = alloc_bf((size_t)512 * 256);
  __bf16* wt_dt_h1 = alloc_bf((size_t)512 * 512);
  __bf16* wt_dt_o  = alloc_bf((size_t)768 * 512);
  __bf16* wt_di_h0 = alloc_bf((size_t)512 * 256);
  __bf16* wt_di_h1 = alloc_bf((size_t)512 * 512);
  __bf16* wt_di_o  = alloc_bf((size_t)512 * 512);
  __bf16* wt_c1    = alloc_bf((size_t)256 * 256);
  __bf16* wt_c2    = alloc_bf((size_t)128 * 256);
  __bf16* cb0b     = alloc_bf((size_t)2048 * 256);
  __bf16* cb1b     = alloc_bf((size_t)1024 * 256);
  __bf16* cb2b     = alloc_bf((size_t)512 * 256);
  __bf16* cb3b     = alloc_bf((size_t)256 * 256);

  float* A1    = alloc_f((size_t)BATCH * 512);
  float* A2    = alloc_f((size_t)BATCH * 512);
  float* Ybt   = alloc_f((size_t)BATCH * 256);
  float* Ybi   = alloc_f((size_t)BATCH * 256);
  float* FUSED = alloc_f((size_t)BATCH * 544);
  float* WN    = alloc_f(3840);
  float* T256  = Ybt;  // Ybt dead after branch pipeline; reuse

  // ---- output carve-up (tuple concatenated flat, float32) ----
  float* out   = (float*)d_out;
  float* oRT   = out;
  float* oRI   = oRT + (size_t)BATCH * 768;
  float* oH    = oRI + (size_t)BATCH * 512;
  float* oZQ   = oH + (size_t)BATCH * 256;
  float* oZQST = oZQ + (size_t)BATCH * 256;
  float* oSIDS = oZQST + (size_t)BATCH * 256;
  float* oRES  = oSIDS + (size_t)BATCH * 4;
  float* oQNT  = oRES + (size_t)BATCH * 1024;
  float* oMASK = oQNT + (size_t)BATCH * 1024;
  float* oCON  = oMASK + 4;

#define PACKT(src, dst, N, K, KP) \
  pack_wt_kernel<<<((N) * (KP) + 255) / 256, 256, 0, stream>>>(src, dst, N, K, KP)

  PACKT(F(24), wt_bt_p0, 512, 768, 768);
  PACKT(F(28), wt_bt_po, 256, 512, 512);
  PACKT(F(18), wt_bt_b1, 512, 256, 256);
  PACKT(F(20), wt_bt_b2, 256, 512, 512);
  PACKT(F(10), wt_bi_p0, 512, 512, 512);
  PACKT(F(14), wt_bi_po, 256, 512, 512);
  PACKT(F(4),  wt_bi_b1, 512, 256, 256);
  PACKT(F(6),  wt_bi_b2, 256, 512, 512);
  PACKT(F(62), wt_f_h0,  512, 513, 544);
  PACKT(F(66), wt_f_h1,  512, 512, 512);
  PACKT(F(70), wt_f_o,   256, 512, 512);
  PACKT(F(52), wt_dt_h0, 512, 256, 256);
  PACKT(F(56), wt_dt_h1, 512, 512, 512);
  PACKT(F(60), wt_dt_o,  768, 512, 512);
  PACKT(F(42), wt_di_h0, 512, 256, 256);
  PACKT(F(46), wt_di_h1, 512, 512, 512);
  PACKT(F(50), wt_di_o,  512, 512, 512);
  PACKT(F(36), wt_c1,    256, 256, 256);
  PACKT(F(38), wt_c2,    128, 256, 256);
  pack_cb_kernel<<<(2048 * 256 + 255) / 256, 256, 0, stream>>>(F(31), cb0b, 2048 * 256);
  pack_cb_kernel<<<(1024 * 256 + 255) / 256, 256, 0, stream>>>(F(32), cb1b, 1024 * 256);
  pack_cb_kernel<<<(512 * 256 + 255) / 256, 256, 0, stream>>>(F(33), cb2b, 512 * 256);
  pack_cb_kernel<<<(256 * 256 + 255) / 256, 256, 0, stream>>>(F(34), cb3b, 256 * 256);
  cb_norm_kernel<<<2048 / 256, 256, 0, stream>>>(F(31), WN, 2048);
  cb_norm_kernel<<<1024 / 256, 256, 0, stream>>>(F(32), WN + 2048, 1024);
  cb_norm_kernel<<<512 / 256, 256, 0, stream>>>(F(33), WN + 3072, 512);
  cb_norm_kernel<<<256 / 256, 256, 0, stream>>>(F(34), WN + 3584, 256);

  const int GB = BATCH / 32;  // 1024 blocks, 32 rows each
#define GEMM(TPW, EPI, Ap, LDA, WT, BIAS, LNG, LNB, RES, LDR, Cp, LDC, COFF, K) \
  gemm_epi<TPW, EPI><<<GB, 256, 0, stream>>>(Ap, LDA, WT, BIAS, LNG, LNB, RES, LDR, Cp, (long)(LDC), (long)(COFF), K)

  // text branch: proj(h0 ln+gelu, out ln) then block (gelu, resid+ln) -> FUSED[:,0:256]
  GEMM(4, 3, text, 768, wt_bt_p0, F(23), F(26), F(25), (const float*)nullptr, 0, A1, 512, 0, 768);
  GEMM(2, 2, A1, 512, wt_bt_po, F(27), F(30), F(29), (const float*)nullptr, 0, Ybt, 256, 0, 512);
  GEMM(4, 1, Ybt, 256, wt_bt_b1, F(17), (const float*)nullptr, (const float*)nullptr, (const float*)nullptr, 0, A2, 512, 0, 256);
  GEMM(2, 2, A2, 512, wt_bt_b2, F(19), F(22), F(21), Ybt, 256, FUSED, 544, 0, 512);
  // image branch -> FUSED[:,256:512]
  GEMM(4, 3, image, 512, wt_bi_p0, F(9), F(12), F(11), (const float*)nullptr, 0, A1, 512, 0, 512);
  GEMM(2, 2, A1, 512, wt_bi_po, F(13), F(16), F(15), (const float*)nullptr, 0, Ybi, 256, 0, 512);
  GEMM(4, 1, Ybi, 256, wt_bi_b1, F(3), (const float*)nullptr, (const float*)nullptr, (const float*)nullptr, 0, A2, 512, 0, 256);
  GEMM(2, 2, A2, 512, wt_bi_b2, F(5), F(8), F(7), Ybi, 256, FUSED, 544, 256, 512);
  fill_fused_kernel<<<(BATCH * 32) / 256, 256, 0, stream>>>(dmask, FUSED);
  // fusion -> h
  GEMM(4, 3, FUSED, 544, wt_f_h0, F(61), F(64), F(63), (const float*)nullptr, 0, A1, 512, 0, 544);
  GEMM(4, 3, A1, 512, wt_f_h1, F(65), F(68), F(67), (const float*)nullptr, 0, A2, 512, 0, 512);
  GEMM(2, 2, A2, 512, wt_f_o, F(69), F(72), F(71), (const float*)nullptr, 0, oH, 256, 0, 512);
  // residual quantization chain (residual slots live in the residuals output)
  rq_level<<<GB, 256, 0, stream>>>(oH, 256, 1, cb0b, F(31), WN, 2048,
                                   oRES, oQNT, oRES + 256, 1024, oZQ, oZQST, oSIDS, act, 0);
  rq_level<<<GB, 256, 0, stream>>>(oRES + 256, 1024, 0, cb1b, F(32), WN + 2048, 1024,
                                   oRES + 256, oQNT + 256, oRES + 512, 1024, oZQ, oZQST, oSIDS, act, 1);
  rq_level<<<GB, 256, 0, stream>>>(oRES + 512, 1024, 0, cb2b, F(33), WN + 3072, 512,
                                   oRES + 512, oQNT + 512, oRES + 768, 1024, oZQ, oZQST, oSIDS, act, 2);
  rq_level<<<GB, 256, 0, stream>>>(oRES + 768, 1024, 0, cb3b, F(34), WN + 3584, 256,
                                   oRES + 768, oQNT + 768, T256, 256, oZQ, oZQST, oSIDS, act, 3);
  write_mask_kernel<<<1, 4, 0, stream>>>(oMASK, act);
  // decoders (input z_q_st)
  GEMM(4, 3, oZQST, 256, wt_dt_h0, F(51), F(54), F(53), (const float*)nullptr, 0, A1, 512, 0, 256);
  GEMM(4, 3, A1, 512, wt_dt_h1, F(55), F(58), F(57), (const float*)nullptr, 0, A2, 512, 0, 512);
  GEMM(6, 0, A2, 512, wt_dt_o, F(59), (const float*)nullptr, (const float*)nullptr, (const float*)nullptr, 0, oRT, 768, 0, 512);
  GEMM(4, 3, oZQST, 256, wt_di_h0, F(41), F(44), F(43), (const float*)nullptr, 0, A1, 512, 0, 256);
  GEMM(4, 3, A1, 512, wt_di_h1, F(45), F(48), F(47), (const float*)nullptr, 0, A2, 512, 0, 512);
  GEMM(4, 0, A2, 512, wt_di_o, F(49), (const float*)nullptr, (const float*)nullptr, (const float*)nullptr, 0, oRI, 512, 0, 512);
  // contrastive head: lin1 -> gelu -> ln (EPI 4), then lin2 -> l2norm (EPI 5)
  GEMM(2, 4, oZQST, 256, wt_c1, F(35), F(40), F(39), (const float*)nullptr, 0, T256, 256, 0, 256);
  GEMM(1, 5, T256, 256, wt_c2, F(37), (const float*)nullptr, (const float*)nullptr, (const float*)nullptr, 0, oCON, 128, 0, 256);
#undef GEMM
#undef PACKT
}
